// AttentionBlock_3264175145238
// MI455X (gfx1250) — matrix operand order repaired and checked by reading, not compiled, not run on hardware
//
#include <hip/hip_runtime.h>
#include <hip/hip_bf16.h>

// Problem constants (match reference)
#define B_   4
#define S_   2048
#define DM_  512
#define DK_  64
#define DV_  64

typedef __bf16 v16bf __attribute__((ext_vector_type(16)));
typedef float  v8f   __attribute__((ext_vector_type(8)));
typedef unsigned int v4u __attribute__((ext_vector_type(4)));
typedef unsigned int v8u __attribute__((ext_vector_type(8)));
typedef unsigned short u16;

union Frag16 {
  v16bf bf;
  uint4 q[2];
  u16   us[16];
};

__device__ __forceinline__ u16 f2bf(float f) {
  unsigned int u = __float_as_uint(f);
  u += 0x7FFFu + ((u >> 16) & 1u);   // round-to-nearest-even
  return (u16)(u >> 16);
}

__device__ __forceinline__ v8f wmma_bf16(v16bf a, v16bf b, v8f c) {
  // D = A(16x32) * B(32x16) + C, f32 accumulate
  return __builtin_amdgcn_wmma_f32_16x16x32_bf16(
      /*neg_a=*/false, a, /*neg_b=*/false, b,
      /*c_mod=*/(short)0, c, /*reuse_a=*/false, /*reuse_b=*/false);
}

// A-fragment (16x32, bf16) from a row-major [M x Kstride] bf16 matrix.
// Lane layout (ISA 7.12.2): lane m = lane%16, hf = lane/16;
//   us[0..7]  = K = kBase + 8*hf + {0..7}
//   us[8..15] = K = kBase + 16 + 8*hf + {0..7}
__device__ __forceinline__ v16bf load_a_bf16(const u16* base, int stride,
                                             int m, int kBase, int hf) {
  Frag16 f;
  const u16* p = base + (size_t)m * stride + kBase + 8 * hf;
  f.q[0] = *(const uint4*)(p);
  f.q[1] = *(const uint4*)(p + 16);
  return f.bf;
}

// B-fragment (32x16, bf16) read from B^T stored row-major [N x Kstride].
// Lane layout: lane n = lane%16, hf = lane/16;
//   us[0..7]  = K = kBase + 8*hf + {0..7}
//   us[8..15] = K = kBase + 16 + 8*hf + {0..7}
__device__ __forceinline__ v16bf load_b_bf16(const u16* baseT, int stride,
                                             int n, int kBase, int hf) {
  Frag16 f;
  const u16* p = baseT + (size_t)n * stride + kBase + 8 * hf;
  f.q[0] = *(const uint4*)(p);
  f.q[1] = *(const uint4*)(p + 16);
  return f.bf;
}

// A-fragment from a row-major f32 matrix, converting to bf16 on the fly.
__device__ __forceinline__ v16bf load_a_f32(const float* base, int stride,
                                            int m, int kBase, int hf) {
  const float* p = base + (size_t)m * stride + kBase + 8 * hf;
  float4 f0 = *(const float4*)(p);
  float4 f1 = *(const float4*)(p + 4);
  float4 f2 = *(const float4*)(p + 16);
  float4 f3 = *(const float4*)(p + 20);
  Frag16 f;
  f.us[0]  = f2bf(f0.x); f.us[1]  = f2bf(f0.y); f.us[2]  = f2bf(f0.z); f.us[3]  = f2bf(f0.w);
  f.us[4]  = f2bf(f1.x); f.us[5]  = f2bf(f1.y); f.us[6]  = f2bf(f1.z); f.us[7]  = f2bf(f1.w);
  f.us[8]  = f2bf(f2.x); f.us[9]  = f2bf(f2.y); f.us[10] = f2bf(f2.z); f.us[11] = f2bf(f2.w);
  f.us[12] = f2bf(f3.x); f.us[13] = f2bf(f3.y); f.us[14] = f2bf(f3.z); f.us[15] = f2bf(f3.w);
  return f.bf;
}

// ---------------------------------------------------------------------------
// TDM: async load of a 16x32 f32 tile (row stride S_ elements) into LDS.
// D# per CDNA5 ISA: 2D tensor -> groups 0+1 only.
// ---------------------------------------------------------------------------
__device__ __forceinline__ void tdm_load_tile_f32(const float* src, unsigned lds_off) {
  unsigned long long ga = (unsigned long long)(uintptr_t)src;
  v4u g0;
  g0[0] = 1u;                                     // count=1, user descriptor
  g0[1] = lds_off;                                // lds_addr (bytes)
  g0[2] = (unsigned)ga;                           // global_addr[31:0]
  g0[3] = (unsigned)((ga >> 32) & 0x1FFFFFFu)     // global_addr[56:32]
        | (2u << 30);                             // type = 2 ("image")
  v8u g1;
  g1[0] = (2u << 16);                             // data_size = 4 bytes
  g1[1] = ((unsigned)S_ & 0xFFFFu) << 16;         // tensor_dim0[15:0]
  g1[2] = ((unsigned)S_ >> 16) | (16u << 16);     // tensor_dim0[31:16] | tensor_dim1[15:0]
  g1[3] = (32u << 16);                            // tensor_dim1[31:16]=0 | tile_dim0=32
  g1[4] = 16u;                                    // tile_dim1=16, tile_dim2=0
  g1[5] = (unsigned)S_;                           // tensor_dim0_stride[31:0]
  g1[6] = 0u;                                     // stride[47:32]=0, dim1_stride lo=0
  g1[7] = 0u;
  asm volatile("tensor_load_to_lds %0, %1" : : "s"(g0), "s"(g1) : "memory");
}

// ---------------------------------------------------------------------------
// Kernel 1: transpose + bf16-convert weights.
//   Wq/Wk/Wv [512x64] -> WT [64x512] bf16 ;  Wfc [64x512] -> WfcT [512x64] bf16
// ---------------------------------------------------------------------------
__global__ void prep_weights(const float* __restrict__ Wq, const float* __restrict__ Wk,
                             const float* __restrict__ Wv, const float* __restrict__ Wfc,
                             u16* __restrict__ WqT, u16* __restrict__ WkT,
                             u16* __restrict__ WvT, u16* __restrict__ WfcT) {
  int idx = blockIdx.x * blockDim.x + threadIdx.x;   // 0 .. 131071
  int mat = idx >> 15;
  int e   = idx & 32767;
  if (mat < 3) {
    int n = e >> 9, k = e & 511;                     // WT[n][k] = W[k][n]
    const float* W = (mat == 0) ? Wq : (mat == 1) ? Wk : Wv;
    u16* WT        = (mat == 0) ? WqT : (mat == 1) ? WkT : WvT;
    WT[e] = f2bf(W[k * DK_ + n]);
  } else {
    int n = e >> 6, k = e & 63;                      // WfcT[n][k] = Wfc[k][n]
    WfcT[e] = f2bf(Wfc[k * DM_ + n]);
  }
}

// ---------------------------------------------------------------------------
// Kernel 2: q/k/v projections.  One wave -> one 16x64 output tile.
//   qp,kp stored row-major bf16 [B*S][64]; vp stored transposed bf16 [B][64][S]
// ---------------------------------------------------------------------------
__global__ void proj_kernel(const float* __restrict__ q, const float* __restrict__ k,
                            const float* __restrict__ v,
                            const u16* __restrict__ WqT, const u16* __restrict__ WkT,
                            const u16* __restrict__ WvT,
                            const float* __restrict__ bq, const float* __restrict__ bk,
                            const float* __restrict__ bv,
                            u16* __restrict__ qp, u16* __restrict__ kp,
                            u16* __restrict__ vpT) {
  const int wid  = threadIdx.x >> 5;
  const int lane = threadIdx.x & 31;
  const int mn   = lane & 15;
  const int hf   = lane >> 4;

  int tile  = blockIdx.x * (blockDim.x >> 5) + wid;  // 0 .. 1535
  int which = tile >> 9;                             // 0=q 1=k 2=v
  int r0    = (tile & 511) * 16;                     // global row (b*S + s)

  const float* X    = (which == 0) ? q   : (which == 1) ? k   : v;
  const u16*   WT   = (which == 0) ? WqT : (which == 1) ? WkT : WvT;
  const float* bias = (which == 0) ? bq  : (which == 1) ? bk  : bv;

  v8f acc[4] = {};
  for (int kc = 0; kc < DM_ / 32; ++kc) {
    int kb = kc * 32;
    v16bf a = load_a_f32(X + (size_t)r0 * DM_, DM_, mn, kb, hf);
#pragma unroll
    for (int g = 0; g < 4; ++g) {
      v16bf b = load_b_bf16(WT, DM_, 16 * g + mn, kb, hf);
      acc[g] = wmma_bf16(a, b, acc[g]);
    }
  }

#pragma unroll
  for (int g = 0; g < 4; ++g) {
    float bv_ = bias[16 * g + mn];
#pragma unroll
    for (int vv = 0; vv < 8; ++vv) {
      int m = vv + 8 * hf;                           // C-layout row
      int r = r0 + m;
      float val = acc[g][vv] + bv_;
      if (which < 2) {
        u16* dst = (which == 0) ? qp : kp;
        dst[(size_t)r * DK_ + 16 * g + mn] = f2bf(val);
      } else {
        int bb = r >> 11, s = r & (S_ - 1);
        vpT[((size_t)bb * DV_ + 16 * g + mn) * S_ + s] = f2bf(val);
      }
    }
  }
}

// ---------------------------------------------------------------------------
// Kernel 3: flash attention.  One wave -> 16 query rows, streams 32 keys/iter.
// g_bias tiles are fetched by the Tensor Data Mover into a double-buffered
// LDS staging area, overlapped with the QK WMMAs (TENSORcnt synchronized).
// ---------------------------------------------------------------------------
__global__ void flash_attn(const u16* __restrict__ qp, const u16* __restrict__ kp,
                           const u16* __restrict__ vpT,
                           const float* __restrict__ g_bias,
                           const unsigned char* __restrict__ mask,
                           const float* __restrict__ tauPtr,
                           u16* __restrict__ ao) {
  __shared__ __align__(16) u16   lds_p[4][16 * 32];       // per-wave P tile (bf16)
  __shared__ __align__(16) float lds_gb[4][2][16 * 32];   // per-wave dbl-buffered bias

  const int wid  = threadIdx.x >> 5;
  const int lane = threadIdx.x & 31;
  const int mn   = lane & 15;
  const int hf   = lane >> 4;

  int tile = blockIdx.x * (blockDim.x >> 5) + wid;   // 0 .. 511
  int b    = tile >> 7;
  int q0   = (tile & 127) * 16;

  // Wave-uniform copies for the TDM descriptor ("s" asm operands).
  int tileU = __builtin_amdgcn_readfirstlane(tile);
  int bU    = tileU >> 7;
  int q0U   = (tileU & 127) * 16;
  const float* gbU = g_bias + ((size_t)bU * S_ + q0U) * S_;
  unsigned lds_buf0 = __builtin_amdgcn_readfirstlane(
      (unsigned)(uintptr_t)(void*)&lds_gb[wid][0][0]);
  unsigned lds_buf1 = __builtin_amdgcn_readfirstlane(
      (unsigned)(uintptr_t)(void*)&lds_gb[wid][1][0]);

  const float tau  = *tauPtr;
  const float coef = 1.0f / (2.0f * tau * tau);
  const float invT = 0.125f;                         // 1 / TEMPERATURE

  const u16* qp_b  = qp  + ((size_t)b * S_ + q0) * DK_;
  const u16* kp_b  = kp  + (size_t)b * S_ * DK_;
  const u16* vpT_b = vpT + (size_t)b * DV_ * S_;
  const unsigned char* mk = mask + ((size_t)b * S_ + q0) * S_;

  // Q fragments, resident for the whole pass (K = 0..31 and 32..63)
  v16bf aQ0 = load_a_bf16(qp_b, DK_, mn, 0,  hf);
  v16bf aQ1 = load_a_bf16(qp_b, DK_, mn, 32, hf);

  float mi[8], li[8];
  v8f acc[4] = {};
#pragma unroll
  for (int vv = 0; vv < 8; ++vv) { mi[vv] = -1e30f; li[vv] = 0.0f; }

  u16* lp = lds_p[wid];
  const int NT = S_ / 32;

  // Prime the TDM pipeline with tile 0.
  tdm_load_tile_f32(gbU, lds_buf0);

  for (int kt = 0; kt < NT; ++kt) {
    const int k0 = kt * 32;

    // Issue next tile's TDM load, then wait for the current tile only.
    if (kt + 1 < NT) {
      tdm_load_tile_f32(gbU + (kt + 1) * 32, ((kt + 1) & 1) ? lds_buf1 : lds_buf0);
      __builtin_amdgcn_s_wait_tensorcnt(1);
    } else {
      __builtin_amdgcn_s_wait_tensorcnt(0);
    }
    asm volatile("" ::: "memory");                   // keep LDS reads below the wait
    const float* ldsB = &lds_gb[wid][kt & 1][0];

    __builtin_prefetch(mk + (size_t)mn * S_ + k0 + 32, 0, 1);

    // ---- S = Q K^T (two 16x16 N-subtiles, K-dim 64 = 2 WMMAs each) ----
    v8f s0 = {}, s1 = {};
    {
      v16bf bK;
      bK = load_b_bf16(kp_b, DK_, k0 + mn,      0,  hf); s0 = wmma_bf16(aQ0, bK, s0);
      bK = load_b_bf16(kp_b, DK_, k0 + mn,      32, hf); s0 = wmma_bf16(aQ1, bK, s0);
      bK = load_b_bf16(kp_b, DK_, k0 + 16 + mn, 0,  hf); s1 = wmma_bf16(aQ0, bK, s1);
      bK = load_b_bf16(kp_b, DK_, k0 + 16 + mn, 32, hf); s1 = wmma_bf16(aQ1, bK, s1);
    }

    // ---- scale + bias (from TDM-staged LDS) + mask ----
#pragma unroll
    for (int vv = 0; vv < 8; ++vv) {
      int m = vv + 8 * hf;
      float b0 = ldsB[m * 32 + mn];
      float b1 = ldsB[m * 32 + 16 + mn];
      size_t moff = (size_t)m * S_ + k0 + mn;
      bool x0 = mk[moff] != 0;
      bool x1 = mk[moff + 16] != 0;
      s0[vv] = x0 ? -1e30f : (s0[vv] * invT + b0 * coef);
      s1[vv] = x1 ? -1e30f : (s1[vv] * invT + b1 * coef);
    }

    // ---- online softmax: row stats via xor-shuffles within 16-lane group ----
    float pr0[8], pr1[8];
#pragma unroll
    for (int vv = 0; vv < 8; ++vv) {
      float rmax = fmaxf(s0[vv], s1[vv]);
#pragma unroll
      for (int off = 1; off < 16; off <<= 1)
        rmax = fmaxf(rmax, __shfl_xor(rmax, off, 32));
      float mnew  = fmaxf(mi[vv], rmax);
      float scale = __expf(mi[vv] - mnew);
      float p0 = __expf(s0[vv] - mnew);
      float p1 = __expf(s1[vv] - mnew);
      float rsum = p0 + p1;
#pragma unroll
      for (int off = 1; off < 16; off <<= 1)
        rsum += __shfl_xor(rsum, off, 32);
      li[vv] = li[vv] * scale + rsum;
      mi[vv] = mnew;
#pragma unroll
      for (int g = 0; g < 4; ++g) acc[g][vv] *= scale;
      pr0[vv] = p0; pr1[vv] = p1;
    }

    // ---- C-layout -> A-layout for P via LDS ----
#pragma unroll
    for (int vv = 0; vv < 8; ++vv) {
      int m = vv + 8 * hf;
      lp[m * 32 + mn]      = f2bf(pr0[vv]);
      lp[m * 32 + 16 + mn] = f2bf(pr1[vv]);
    }
    __syncthreads();
    Frag16 aP;
    {
      const u16* p = lp + mn * 32 + 8 * hf;
      aP.q[0] = *(const uint4*)(p);
      aP.q[1] = *(const uint4*)(p + 16);
    }
    __syncthreads();

    // ---- O += P V  (4 d-subtiles of 16) ----
#pragma unroll
    for (int g = 0; g < 4; ++g) {
      v16bf bV = load_b_bf16(vpT_b, S_, 16 * g + mn, k0, hf);
      acc[g] = wmma_bf16(aP.bf, bV, acc[g]);
    }
  }

  // ---- finalize: O / l, store bf16 row-major [B*S][64] ----
#pragma unroll
  for (int g = 0; g < 4; ++g)
#pragma unroll
    for (int vv = 0; vv < 8; ++vv) {
      int m = vv + 8 * hf;
      float o = acc[g][vv] / li[vv];
      ao[((size_t)(b * S_ + q0 + m)) * DV_ + 16 * g + mn] = f2bf(o);
    }
}

// ---------------------------------------------------------------------------
// Kernel 4: output projection  out = ao @ Wfc + bfc  (f32 result)
//   One wave -> 16 rows x 64 cols.
// ---------------------------------------------------------------------------
__global__ void final_proj(const u16* __restrict__ ao, const u16* __restrict__ WfcT,
                           const float* __restrict__ bfc, float* __restrict__ out) {
  const int wid  = threadIdx.x >> 5;
  const int lane = threadIdx.x & 31;
  const int mn   = lane & 15;
  const int hf   = lane >> 4;

  int w  = blockIdx.x * (blockDim.x >> 5) + wid;     // 0 .. 4095
  int r0 = (w >> 3) * 16;
  int c0 = (w & 7) * 64;

  v16bf a0 = load_a_bf16(ao + (size_t)r0 * DV_, DV_, mn, 0,  hf);
  v16bf a1 = load_a_bf16(ao + (size_t)r0 * DV_, DV_, mn, 32, hf);

#pragma unroll
  for (int g = 0; g < 4; ++g) {
    int n = c0 + 16 * g + mn;
    v16bf b0 = load_b_bf16(WfcT, DV_, n, 0,  hf);
    v16bf b1 = load_b_bf16(WfcT, DV_, n, 32, hf);
    v8f acc = {};
    acc = wmma_bf16(a0, b0, acc);
    acc = wmma_bf16(a1, b1, acc);
    float bias_v = bfc[n];
#pragma unroll
    for (int vv = 0; vv < 8; ++vv) {
      int m = vv + 8 * hf;
      out[(size_t)(r0 + m) * DM_ + n] = acc[vv] + bias_v;
    }
  }
}

// ---------------------------------------------------------------------------
extern "C" void kernel_launch(void* const* d_in, const int* in_sizes, int n_in,
                              void* d_out, int out_size, void* d_ws, size_t ws_size,
                              hipStream_t stream) {
  (void)in_sizes; (void)n_in; (void)out_size; (void)ws_size;

  const float*         q      = (const float*)d_in[0];
  const float*         k      = (const float*)d_in[1];
  const float*         v      = (const float*)d_in[2];
  const float*         g_bias = (const float*)d_in[3];
  const unsigned char* mask   = (const unsigned char*)d_in[4];
  const float*         Wq     = (const float*)d_in[5];
  const float*         bq     = (const float*)d_in[6];
  const float*         Wk     = (const float*)d_in[7];
  const float*         bk     = (const float*)d_in[8];
  const float*         Wv     = (const float*)d_in[9];
  const float*         bv     = (const float*)d_in[10];
  const float*         Wfc    = (const float*)d_in[11];
  const float*         bfc    = (const float*)d_in[12];
  const float*         tau    = (const float*)d_in[13];
  float* out = (float*)d_out;

  // Workspace layout (bf16 / u16 units)
  u16* W = (u16*)d_ws;
  const size_t NROW = (size_t)B_ * S_;               // 8192
  u16* qp   = W;                                     // [8192][64]
  u16* kp   = qp  + NROW * DK_;                      // [8192][64]
  u16* vpT  = kp  + NROW * DK_;                      // [B][64][S]
  u16* ao   = vpT + NROW * DV_;                      // [8192][64]
  u16* WqT  = ao  + NROW * DV_;                      // [64][512]
  u16* WkT  = WqT + (size_t)DK_ * DM_;
  u16* WvT  = WkT + (size_t)DK_ * DM_;
  u16* WfcT = WvT + (size_t)DK_ * DM_;               // [512][64]

  // 1) weight transpose/convert: 4 * 32768 elements
  prep_weights<<<512, 256, 0, stream>>>(Wq, Wk, Wv, Wfc, WqT, WkT, WvT, WfcT);

  // 2) projections: 3 * 512 tiles, 4 waves/block
  proj_kernel<<<384, 128, 0, stream>>>(q, k, v, WqT, WkT, WvT, bq, bk, bv,
                                       qp, kp, vpT);

  // 3) flash attention: 512 tiles, 4 waves/block
  flash_attn<<<128, 128, 0, stream>>>(qp, kp, vpT, g_bias, mask, tau, ao);

  // 4) output projection: 4096 waves, 4 waves/block
  final_proj<<<1024, 128, 0, stream>>>(ao, WfcT, bfc, out);
}
